// FFModule_57269093925317
// MI455X (gfx1250) — matrix operand order repaired and checked by reading, not compiled, not run on hardware
//
#include <hip/hip_runtime.h>
#include <hip/hip_bf16.h>
#include <math.h>

// ---------------------------------------------------------------------------
// Dims / offsets
// ---------------------------------------------------------------------------
#define Bn   2
#define Cn   64
#define Hn   512
#define Wn   512
#define HWn  (Hn*Wn)            // 262144
#define Tn   16
#define GHn  (Hn/Tn)            // 32
#define GWn  (Wn/Tn)            // 32
#define Kn   (GHn*GWn)          // 1024
#define KMAXn 256
#define HIDn (2*Cn)             // 128

// d_out element offsets (float)
#define OFF_HEAVY  ((size_t)0)
#define OFF_DETAIL ((size_t)Bn*Cn*HWn)                  // 33554432
#define OFF_ALPHA  (OFF_DETAIL + (size_t)Bn*Cn*HWn)     // 67108864
#define OFF_PROB   (OFF_ALPHA + (size_t)Bn*HWn)         // 67633152
#define OFF_GATES  (OFF_PROB + (size_t)Bn*Kn)           // 67635200
#define OFF_COST   (OFF_GATES + (size_t)Bn*Kn)          // 67637248
#define OFF_LOSS   (OFF_COST + 1)

typedef __attribute__((ext_vector_type(16))) __bf16    v16bf;
typedef __attribute__((ext_vector_type(8)))  float     v8f;
typedef __attribute__((ext_vector_type(8)))  unsigned  v8u;

#define LOG2E 1.4426950408889634f

// branch-free fast transcendentals (v_exp_f32 / v_rcp_f32)
__device__ __forceinline__ float fast_exp(float x) {
    return __builtin_amdgcn_exp2f(x * LOG2E);
}
__device__ __forceinline__ float fast_sigmoid(float z) {
    return __builtin_amdgcn_rcpf(1.0f + fast_exp(-z));
}
// gelu(v) = 0.5 v (1 + tanh(g)) = v * sigmoid(2g),  g = k0 (v + k1 v^3)
__device__ __forceinline__ float gelu_fast(float v) {
    const float k0 = 0.7978845608028654f;   // sqrt(2/pi)
    const float k1 = 0.044715f;
    float g = k0 * (v + k1 * v * v * v);
    return v * fast_sigmoid(2.0f * g);
}
__device__ __forceinline__ float sigmoid_accurate(float v) {
    return 1.0f / (1.0f + expf(-v));
}
__device__ __forceinline__ unsigned short f2bf_bits(float f) {
    return __builtin_bit_cast(unsigned short, (__bf16)f);
}
// packed f32x2 -> bf16x2 in one u32 (one v_cvt_pk if available)
__device__ __forceinline__ unsigned pk_bf16(float f0, float f1) {
#if __has_builtin(__builtin_amdgcn_cvt_pk_bf16_f32)
    auto pk = __builtin_amdgcn_cvt_pk_bf16_f32(f0, f1);
    return __builtin_bit_cast(unsigned, pk);
#else
    return (unsigned)f2bf_bits(f0) | ((unsigned)f2bf_bits(f1) << 16);
#endif
}

// ---------------------------------------------------------------------------
// Kernel 0: pack W1 [64x128] and W2 [128x64] (f32 row-major) into bf16
// K-pair-interleaved u32 words: Wp[k/2][n] = {bf16 W[k][n], bf16 W[k+1][n]}
// ---------------------------------------------------------------------------
__global__ void pack_weights_kernel(const float* __restrict__ W1,
                                    const float* __restrict__ W2,
                                    unsigned* __restrict__ w1p,
                                    unsigned* __restrict__ w2p) {
    int tid = blockIdx.x * blockDim.x + threadIdx.x;
    int nthr = gridDim.x * blockDim.x;
    for (int i = tid; i < (Cn/2) * HIDn; i += nthr) {       // 32 x 128
        int kp = i >> 7, n = i & 127;
        w1p[i] = pk_bf16(W1[(2*kp) * HIDn + n], W1[(2*kp+1) * HIDn + n]);
    }
    for (int i = tid; i < (HIDn/2) * Cn; i += nthr) {       // 64 x 64
        int kp = i >> 6, n = i & 63;
        w2p[i] = pk_bf16(W2[(2*kp) * Cn + n], W2[(2*kp+1) * Cn + n]);
    }
}

// ---------------------------------------------------------------------------
// Kernel 1: gate probabilities / STE gates, top-KMAX selection mask,
// expected_cost and budget_loss. One block of 1024 threads, loops over B.
// ---------------------------------------------------------------------------
__global__ __launch_bounds__(Kn) void gate_kernel(const float* __restrict__ u,
                                                  float* __restrict__ prob,
                                                  float* __restrict__ gates,
                                                  float* __restrict__ cost_out,
                                                  float* __restrict__ loss_out,
                                                  int* __restrict__ maskb) {
    __shared__ float su[Kn];
    __shared__ float red[Kn];
    int k = threadIdx.x;
    float total = 0.0f;
    for (int b = 0; b < Bn; b++) {
        float uv = u[b * Kn + k];
        float p  = sigmoid_accurate(uv);
        prob[b * Kn + k] = p;
        float hard = (uv >= 0.0f) ? 1.0f : 0.0f;
        gates[b * Kn + k] = (hard + p) - p;                 // STE forward
        float mk = (uv >= 0.0f) ? uv : -1e30f;
        su[k] = mk;
        __syncthreads();
        int cnt = 0;
        for (int j = 0; j < Kn; j++) {
            float mj = su[j];
            cnt += ((mj > mk) || (mj == mk && j < k)) ? 1 : 0;
        }
        maskb[b * Kn + k] = (mk > -1e29f && cnt < KMAXn) ? 1 : 0;
        red[k] = p * 1.0f + (1.0f - p) * 0.1f;              // C_HEAVY/C_CHEAP
        __syncthreads();
        for (int s = Kn / 2; s > 0; s >>= 1) {
            if (k < s) red[k] += red[k + s];
            __syncthreads();
        }
        if (k == 0) total += red[0];
        __syncthreads();
    }
    if (k == 0) {
        *cost_out = total;
        float ex = total - 512.0f * (float)Bn;              // BUDGET_TOTAL*B
        *loss_out = ex > 0.0f ? ex : 0.0f;                  // MU = 1
    }
}

// ---------------------------------------------------------------------------
// Kernel 2: heavy path. One wave32 per aligned 16-pixel group (mask-uniform,
// so EXEC stays all-ones around the WMMAs). All global<->LDS tile traffic is
// float4 (b128). 4 waves / 128-thread block; LDS ~48.5KB.
// ---------------------------------------------------------------------------
__global__ __launch_bounds__(128) void heavy_kernel(
        const float* __restrict__ x,
        const unsigned* __restrict__ w1p,   // [32][128] packed bf16 pairs
        const unsigned* __restrict__ w2p,   // [64][64]
        const float* __restrict__ b1,
        const float* __restrict__ b2,
        const float* __restrict__ Wa,
        const float* __restrict__ ba,
        const int* __restrict__ maskb,
        float* __restrict__ heavy,
        float* __restrict__ detail,
        float* __restrict__ alpha_out) {
    __shared__ float xs_all[4][Cn][16];
    __shared__ float hs_all[4][HIDn][16];
    __shared__ float als_all[4][16];

    const int wave = threadIdx.x >> 5;
    const int lane = threadIdx.x & 31;
    const int gid  = blockIdx.x * 4 + wave;     // 32768 groups total
    const int b    = gid >> 14;                 // 16384 groups per sample
    const int g    = gid & 16383;
    const int p0   = g << 4;                    // hw index of first pixel
    const int hh   = p0 >> 9;
    const int ww   = p0 & (Wn - 1);
    const int sel  = maskb[b * Kn + (hh >> 4) * GWn + (ww >> 4)];
    const size_t xbase = (size_t)b * Cn * HWn + (size_t)p0;

    if (!sel) {                                 // cheap path: b128 copy/zero
        const float4 z4 = {0.f, 0.f, 0.f, 0.f};
        #pragma unroll
        for (int i = 0; i < 8; i++) {
            int idx = i * 32 + lane;
            int c = idx >> 2, q = idx & 3;
            size_t a = xbase + (size_t)c * HWn + q * 4;
            float4 xv = *(const float4*)&x[a];
            *(float4*)&heavy[a]  = xv;
            *(float4*)&detail[a] = z4;
        }
        if (lane < 16) alpha_out[(size_t)b * HWn + p0 + lane] = 0.0f;
        return;
    }

    float (*xw)[16] = xs_all[wave];
    float (*hw)[16] = hs_all[wave];

    // stage x tile: 64 channels x 16 pixels, b128 both sides
    #pragma unroll
    for (int i = 0; i < 8; i++) {
        int idx = i * 32 + lane;
        int c = idx >> 2, q = idx & 3;
        *(float4*)&xw[c][q * 4] =
            *(const float4*)&x[xbase + (size_t)c * HWn + q * 4];
    }

    const int m   = lane & 15;          // A: row M / B,C: column N
    const int kh  = (lane >> 4) * 8;    // A-frag K offset within each 16-block
    const int bkh = (lane >> 4) * 8;    // B-frag K offset
    const int mh  = (lane >> 4) * 8;    // C-frag M offset

    // ---- layer 1: [16x64] x [64x128] + b1, gelu -> h in LDS ----
    for (int nt = 0; nt < 8; nt++) {
        v8f acc = {0.f,0.f,0.f,0.f,0.f,0.f,0.f,0.f};
        int n = nt * 16 + m;
        #pragma unroll
        for (int kb = 0; kb < Cn; kb += 32) {
            v8u au;
            #pragma unroll
            for (int vg = 0; vg < 8; vg++) {
                int k = kb + ((vg < 4) ? 0 : 16) + kh + 2 * (vg & 3);
                au[vg] = pk_bf16(xw[k][m], xw[k+1][m]);
            }
            v16bf af = __builtin_bit_cast(v16bf, au);
            v8u bu;
            #pragma unroll
            for (int vg = 0; vg < 8; vg++) {
                int k = kb + ((vg < 4) ? 0 : 16) + bkh + 2 * (vg & 3);
                bu[vg] = w1p[(k >> 1) * HIDn + n];
            }
            v16bf bf = __builtin_bit_cast(v16bf, bu);
            acc = __builtin_amdgcn_wmma_f32_16x16x32_bf16(
                    false, af, false, bf, (short)0, acc, false, false);
        }
        float bb = b1[n];
        #pragma unroll
        for (int r = 0; r < 8; r++)
            hw[n][r + mh] = gelu_fast(acc[r] + bb);
    }

    // ---- layer 2: [16x128] x [128x64], keep all 4 N-tiles in registers ----
    v8f d2[4];
    #pragma unroll
    for (int n2 = 0; n2 < 4; n2++) {
        v8f acc = {0.f,0.f,0.f,0.f,0.f,0.f,0.f,0.f};
        int n = n2 * 16 + m;
        #pragma unroll
        for (int k2 = 0; k2 < 4; k2++) {
            int kb = k2 * 32;
            v8u au;
            #pragma unroll
            for (int vg = 0; vg < 8; vg++) {
                int k = kb + ((vg < 4) ? 0 : 16) + kh + 2 * (vg & 3);
                au[vg] = pk_bf16(hw[k][m], hw[k+1][m]);
            }
            v16bf af = __builtin_bit_cast(v16bf, au);
            v8u bu;
            #pragma unroll
            for (int vg = 0; vg < 8; vg++) {
                int k = kb + ((vg < 4) ? 0 : 16) + bkh + 2 * (vg & 3);
                bu[vg] = w2p[(k >> 1) * Cn + n];
            }
            v16bf bf = __builtin_bit_cast(v16bf, bu);
            acc = __builtin_amdgcn_wmma_f32_16x16x32_bf16(
                    false, af, false, bf, (short)0, acc, false, false);
        }
        d2[n2] = acc;
    }

    // ---- alpha: per-pixel sigmoid(x . Wa + ba), lanes 0..15 ----
    if (lane < 16) {
        float s = ba[0];
        for (int c = 0; c < Cn; c++) s += xw[c][lane] * Wa[c];
        als_all[wave][lane] = fast_sigmoid(s);
    }

    // ---- stage detail (+b2) into LDS rows 0..63 (h no longer needed) ----
    #pragma unroll
    for (int n2 = 0; n2 < 4; n2++) {
        int c = n2 * 16 + m;
        float bb = b2[c];
        #pragma unroll
        for (int r = 0; r < 8; r++)
            hw[c][r + mh] = d2[n2][r] + bb;
    }

    // ---- b128 stores: heavy = x + alpha*detail, detail_out, alpha_out ----
    #pragma unroll
    for (int i = 0; i < 8; i++) {
        int idx = i * 32 + lane;
        int c = idx >> 2, q = idx & 3;
        size_t a = xbase + (size_t)c * HWn + q * 4;
        float4 dv = *(const float4*)&hw[c][q * 4];
        float4 xv = *(const float4*)&xw[c][q * 4];
        float4 av = *(const float4*)&als_all[wave][q * 4];
        float4 hv;
        hv.x = xv.x + av.x * dv.x;
        hv.y = xv.y + av.y * dv.y;
        hv.z = xv.z + av.z * dv.z;
        hv.w = xv.w + av.w * dv.w;
        *(float4*)&heavy[a]  = hv;
        *(float4*)&detail[a] = dv;
    }
    if (lane < 16)
        alpha_out[(size_t)b * HWn + p0 + lane] = als_all[wave][lane];
}

// ---------------------------------------------------------------------------
extern "C" void kernel_launch(void* const* d_in, const int* in_sizes, int n_in,
                              void* d_out, int out_size, void* d_ws, size_t ws_size,
                              hipStream_t stream) {
    const float* x  = (const float*)d_in[0];
    const float* u  = (const float*)d_in[1];
    const float* W1 = (const float*)d_in[2];
    const float* b1 = (const float*)d_in[3];
    const float* W2 = (const float*)d_in[4];
    const float* b2 = (const float*)d_in[5];
    const float* Wa = (const float*)d_in[6];
    const float* ba = (const float*)d_in[7];

    float* out      = (float*)d_out;
    float* heavy    = out + OFF_HEAVY;
    float* detail   = out + OFF_DETAIL;
    float* alpha    = out + OFF_ALPHA;
    float* prob     = out + OFF_PROB;
    float* gates    = out + OFF_GATES;
    float* cost     = out + OFF_COST;
    float* loss     = out + OFF_LOSS;

    // workspace: mask[2048] int, w1p[4096] u32, w2p[4096] u32
    int*      maskb = (int*)d_ws;
    unsigned* w1p   = (unsigned*)d_ws + 2048;
    unsigned* w2p   = (unsigned*)d_ws + 2048 + 4096;

    pack_weights_kernel<<<dim3(16), dim3(256), 0, stream>>>(W1, W2, w1p, w2p);
    gate_kernel<<<dim3(1), dim3(Kn), 0, stream>>>(u, prob, gates, cost, loss, maskb);
    // 32768 groups of 16 pixels, 4 waves per 128-thread block
    heavy_kernel<<<dim3(8192), dim3(128), 0, stream>>>(
        x, w1p, w2p, b1, b2, Wa, ba, maskb, heavy, detail, alpha);
}
